// ChannelAttention_72258529788552
// MI455X (gfx1250) — matrix operand order repaired and checked by reading, not compiled, not run on hardware
//
#include <hip/hip_runtime.h>
#include <hip/hip_bf16.h>

#define CCH 256
#define TTT 2048
#define NBB 16
#define NL  5
#define BN_EPS 1e-5f

typedef __attribute__((ext_vector_type(16))) __bf16 v16bf;
typedef __attribute__((ext_vector_type(8)))  __bf16 v8bf;
typedef __attribute__((ext_vector_type(8)))  float  v8f;

// K index inside a 16x32 bf16 A fragment for halfword i (0..15) and lane-half h (ISA 7.12.2)
__device__ __forceinline__ int kidxA(int i, int h) {
    int p = i >> 1, w = i & 1;
    int base = (p < 4) ? (2 * p) : (16 + 2 * (p - 4));
    return base + 8 * h + w;
}

// ---------------- prep: qw column sums (coalesced, one block per lag), qb sums, folded BN ----------------
__global__ __launch_bounds__(256) void prep1_kernel(
    const float* __restrict__ q_w, const float* __restrict__ q_b,
    const float* __restrict__ v_b, const float* __restrict__ gmm,
    const float* __restrict__ bet, const float* __restrict__ mu,
    const float* __restrict__ var,
    float* __restrict__ qws, float* __restrict__ qbs,
    float* __restrict__ vsc, float* __restrict__ vsh) {
    const int blk = blockIdx.x;
    const int tid = threadIdx.x;
    if (blk < NL) {
        // column sum over i of q_w[l, i, j]; thread = j -> coalesced per row
        const float* wp = q_w + (size_t)blk * CCH * CCH + tid;
        float s = 0.f;
        for (int i = 0; i < CCH; ++i) s += wp[(size_t)i * CCH];
        qws[blk * CCH + tid] = s;
    } else {
        if (tid < NL) {
            float s = 0.f;
            for (int i = 0; i < CCH; ++i) s += q_b[tid * CCH + i];
            qbs[tid] = s;
        }
        if (tid < CCH) {
            float sc = gmm[tid] * rsqrtf(var[tid] + BN_EPS);
            vsc[tid] = sc;
            vsh[tid] = (v_b[tid] - mu[tid]) * sc + bet[tid];
        }
    }
}

// ---------------- pack k_w / v_w into per-lane WMMA A-fragment order ----------------
#define NSE (40 * 16 * 32 * 16) // 327680 scores A elements
#define NVE (8  * 16 * 32 * 16) // 65536  v A elements
__global__ __launch_bounds__(256) void pack_kernel(
    const float* __restrict__ k_w, const float* __restrict__ v_w,
    __bf16* __restrict__ apS, __bf16* __restrict__ apV) {
    int idx = blockIdx.x * 256 + threadIdx.x;
    if (idx < NSE) {
        int i = idx & 15, lane = (idx >> 4) & 31, mt = (idx >> 9) & 15, s = idx >> 13;
        int c = mt * 16 + (lane & 15), h = lane >> 4;
        int K = s * 32 + kidxA(i, h);
        int l = K >> 8, j = K & 255;
        apS[idx] = (__bf16)k_w[((size_t)l * CCH + c) * CCH + j];
    } else if (idx < NSE + NVE) {
        int e = idx - NSE;
        int i = e & 15, lane = (e >> 4) & 31, mt = (e >> 9) & 15, s = e >> 13;
        int c = mt * 16 + (lane & 15), h = lane >> 4;
        int K = s * 32 + kidxA(i, h);
        apV[e] = (__bf16)v_w[(size_t)c * CCH + K];
    }
}

// ---------------- qsum[b][l][t]: single pass over x, all 5 lags at once ----------------
__global__ __launch_bounds__(256) void qsum_kernel(
    const float* __restrict__ x, const float* __restrict__ qws,
    const float* __restrict__ qbs, float* __restrict__ qsum) {
    int blk = blockIdx.x;          // b*8 + tchunk
    int tch = blk & 7, b = blk >> 3;
    int t = tch * 256 + threadIdx.x;
    const float* xb = x + (size_t)b * CCH * TTT + t;
    float s0 = qbs[0], s1 = qbs[1], s2 = qbs[2], s3 = qbs[3], s4 = qbs[4];
    for (int j = 0; j < CCH; ++j) {
        float xv = xb[(size_t)j * TTT];
        s0 += qws[j]           * xv;
        s1 += qws[CCH + j]     * xv;
        s2 += qws[2 * CCH + j] * xv;
        s3 += qws[3 * CCH + j] * xv;
        s4 += qws[4 * CCH + j] * xv;
    }
    float* qp = qsum + (size_t)b * NL * TTT + t;
    qp[0]       = s0;
    qp[TTT]     = s1;
    qp[2 * TTT] = s2;
    qp[3 * TTT] = s3;
    qp[4 * TTT] = s4;
}

// ---------------- fused scores-GEMM + softmax + v-GEMM + BN/PReLU + multiply ----------------
#define XT_H (36 * 264)
#define XS_H (NL * 32 * 264)
#define SMEM_BYTES ((XT_H + XS_H) * 2 + (NL * 32 + NL * CCH + 32 * 17 + 32) * 4)

__global__ __launch_bounds__(256) void main_kernel(
    const float* __restrict__ x, const float* __restrict__ k_b,
    const float* __restrict__ alpha_p, const float* __restrict__ qsum,
    const __bf16* __restrict__ apS, const __bf16* __restrict__ apV,
    const float* __restrict__ vsc, const float* __restrict__ vsh,
    float* __restrict__ out) {
    extern __shared__ char smem[];
    __bf16* xt  = (__bf16*)smem;       // [36][264] raw x (bf16, transposed), t = t0-4+tt
    __bf16* xs  = xt + XT_H;           // [5][32][264] lag-shifted * qsum-scaled x
    float*  qsm = (float*)(xs + XS_H); // [5][32]
    float*  kb  = qsm + NL * 32;       // [5][256]
    float*  red = kb + NL * CCH;       // [32][17]
    float*  col = red + 32 * 17;       // [32]

    const int tid = threadIdx.x;
    const int b  = blockIdx.y;
    const int t0 = blockIdx.x * 32;

    // stage raw x window
    {
        const float* xrow = x + ((size_t)b * CCH + tid) * TTT + (t0 - 4);
        #pragma unroll
        for (int q4 = 0; q4 < 9; ++q4) {
            int tt = q4 * 4;
            float v0, v1, v2, v3;
            if (t0 >= 4) {
                float4 f = *(const float4*)(xrow + tt);
                v0 = f.x; v1 = f.y; v2 = f.z; v3 = f.w;
            } else {
                v0 = (t0 - 4 + tt + 0 >= 0) ? xrow[tt + 0] : 0.f;
                v1 = (t0 - 4 + tt + 1 >= 0) ? xrow[tt + 1] : 0.f;
                v2 = (t0 - 4 + tt + 2 >= 0) ? xrow[tt + 2] : 0.f;
                v3 = (t0 - 4 + tt + 3 >= 0) ? xrow[tt + 3] : 0.f;
            }
            xt[(tt + 0) * 264 + tid] = (__bf16)v0;
            xt[(tt + 1) * 264 + tid] = (__bf16)v1;
            xt[(tt + 2) * 264 + tid] = (__bf16)v2;
            xt[(tt + 3) * 264 + tid] = (__bf16)v3;
        }
    }
    if (tid < NL * 32) {
        int l = tid >> 5, tc = tid & 31;
        int tg = t0 + tc - l;
        qsm[tid] = (tg >= 0) ? qsum[((size_t)b * NL + l) * TTT + tg] : 0.f;
    }
    for (int e = tid; e < NL * CCH; e += 256) kb[e] = k_b[e];
    __syncthreads();

    // build shifted/scaled copies (once per block, shared by all waves)
    for (int e = tid; e < NL * 32 * CCH; e += 256) {
        int l = e >> 13;
        int r = e & 8191;
        int tc = r >> 8;
        int j = r & 255;
        float v = (float)xt[(tc + 4 - l) * 264 + j] * qsm[l * 32 + tc];
        xs[(l * 32 + tc) * 264 + j] = (__bf16)v;
    }
    __syncthreads();

    const int lane = tid & 31;
    const int wv   = tid >> 5;
    const int n    = lane & 15;
    const int h    = lane >> 4;

    v8f accS[2][2];
    {
        v8f z = {0.f, 0.f, 0.f, 0.f, 0.f, 0.f, 0.f, 0.f};
        accS[0][0] = z; accS[0][1] = z; accS[1][0] = z; accS[1][1] = z;
    }

    // scores GEMM: M=256 (16 tiles), N=32 (2 tiles), K=1280 (40 steps of 32)
    for (int s = 0; s < 40; ++s) {
        int lag = s >> 3;
        int jb  = (s & 7) << 5;
        const __bf16* ap = apS + ((size_t)((s * 16 + 2 * wv) * 32 + lane) << 4);
        v16bf A0 = *(const v16bf*)ap;
        v16bf A1 = *(const v16bf*)(ap + 512);
        const __bf16* bbase = xs + (lag * 32 + n) * 264 + jb + 8 * h;
        #pragma unroll
        for (int nt = 0; nt < 2; ++nt) {
            const __bf16* bp = bbase + nt * 16 * 264;
            v8bf lo = *(const v8bf*)bp;
            v8bf hi = *(const v8bf*)(bp + 16);
            v16bf Bf = __builtin_shufflevector(lo, hi, 0,1,2,3,4,5,6,7,8,9,10,11,12,13,14,15);
            accS[0][nt] = __builtin_amdgcn_wmma_f32_16x16x32_bf16(false, A0, false, Bf, (short)0, accS[0][nt], false, false);
            accS[1][nt] = __builtin_amdgcn_wmma_f32_16x16x32_bf16(false, A1, false, Bf, (short)0, accS[1][nt], false, false);
        }
    }

    // bias: + sum_l k_b[l,row] * qsum_sh[l,t]
    #pragma unroll
    for (int mi = 0; mi < 2; ++mi) {
        #pragma unroll
        for (int r = 0; r < 8; ++r) {
            int row = (2 * wv + mi) * 16 + r + 8 * h;
            float b0 = 0.f, b1 = 0.f;
            #pragma unroll
            for (int l = 0; l < NL; ++l) {
                float kv = kb[l * CCH + row];
                b0 += kv * qsm[l * 32 + n];
                b1 += kv * qsm[l * 32 + 16 + n];
            }
            accS[mi][0][r] += b0;
            accS[mi][1][r] += b1;
        }
    }

    // softmax over channels (cross-wave via LDS)
    float lm0 = -3.0e38f, lm1 = -3.0e38f;
    #pragma unroll
    for (int mi = 0; mi < 2; ++mi)
        #pragma unroll
        for (int r = 0; r < 8; ++r) {
            lm0 = fmaxf(lm0, accS[mi][0][r]);
            lm1 = fmaxf(lm1, accS[mi][1][r]);
        }
    red[n * 17 + (wv * 2 + h)]        = lm0;
    red[(16 + n) * 17 + (wv * 2 + h)] = lm1;
    __syncthreads();
    if (tid < 32) {
        float m = -3.0e38f;
        for (int k2 = 0; k2 < 16; ++k2) m = fmaxf(m, red[tid * 17 + k2]);
        col[tid] = m;
    }
    __syncthreads();
    float cm0 = col[n], cm1 = col[16 + n];
    float ls0 = 0.f, ls1 = 0.f;
    #pragma unroll
    for (int mi = 0; mi < 2; ++mi)
        #pragma unroll
        for (int r = 0; r < 8; ++r) {
            float e0 = __expf(accS[mi][0][r] - cm0);
            float e1 = __expf(accS[mi][1][r] - cm1);
            accS[mi][0][r] = e0; accS[mi][1][r] = e1;
            ls0 += e0; ls1 += e1;
        }
    red[n * 17 + (wv * 2 + h)]        = ls0;
    red[(16 + n) * 17 + (wv * 2 + h)] = ls1;
    __syncthreads();
    if (tid < 32) {
        float sm = 0.f;
        for (int k2 = 0; k2 < 16; ++k2) sm += red[tid * 17 + k2];
        col[tid] = sm;
    }
    __syncthreads();
    float rs0 = 1.f / col[n], rs1 = 1.f / col[16 + n];

    // v GEMM: K=256 (8 steps), B from raw xt (unshifted: tt = tc + 4)
    v8f accV[2][2];
    {
        v8f z = {0.f, 0.f, 0.f, 0.f, 0.f, 0.f, 0.f, 0.f};
        accV[0][0] = z; accV[0][1] = z; accV[1][0] = z; accV[1][1] = z;
    }
    for (int s = 0; s < 8; ++s) {
        int jb = s << 5;
        const __bf16* ap = apV + ((size_t)((s * 16 + 2 * wv) * 32 + lane) << 4);
        v16bf A0 = *(const v16bf*)ap;
        v16bf A1 = *(const v16bf*)(ap + 512);
        const __bf16* bbase = xt + (n + 4) * 264 + jb + 8 * h;
        #pragma unroll
        for (int nt = 0; nt < 2; ++nt) {
            const __bf16* bp = bbase + nt * 16 * 264;
            v8bf lo = *(const v8bf*)bp;
            v8bf hi = *(const v8bf*)(bp + 16);
            v16bf Bf = __builtin_shufflevector(lo, hi, 0,1,2,3,4,5,6,7,8,9,10,11,12,13,14,15);
            accV[0][nt] = __builtin_amdgcn_wmma_f32_16x16x32_bf16(false, A0, false, Bf, (short)0, accV[0][nt], false, false);
            accV[1][nt] = __builtin_amdgcn_wmma_f32_16x16x32_bf16(false, A1, false, Bf, (short)0, accV[1][nt], false, false);
        }
    }

    // BN + PReLU + multiply by softmax weights, store
    float alpha = alpha_p[0];
    #pragma unroll
    for (int mi = 0; mi < 2; ++mi) {
        #pragma unroll
        for (int r = 0; r < 8; ++r) {
            int row = (2 * wv + mi) * 16 + r + 8 * h;
            float sc = vsc[row], sf = vsh[row];
            float vv0 = accV[mi][0][r] * sc + sf;
            float vv1 = accV[mi][1][r] * sc + sf;
            vv0 = (vv0 >= 0.f) ? vv0 : alpha * vv0;
            vv1 = (vv1 >= 0.f) ? vv1 : alpha * vv1;
            float* op = out + ((size_t)b * CCH + row) * TTT + t0;
            op[n]      = vv0 * accS[mi][0][r] * rs0;
            op[16 + n] = vv1 * accS[mi][1][r] * rs1;
        }
    }
}

extern "C" void kernel_launch(void* const* d_in, const int* in_sizes, int n_in,
                              void* d_out, int out_size, void* d_ws, size_t ws_size,
                              hipStream_t stream) {
    (void)in_sizes; (void)n_in; (void)out_size; (void)ws_size;
    const float* x     = (const float*)d_in[0];
    const float* q_w   = (const float*)d_in[1];
    const float* q_b   = (const float*)d_in[2];
    const float* k_w   = (const float*)d_in[3];
    const float* k_b   = (const float*)d_in[4];
    const float* v_w   = (const float*)d_in[5];
    const float* v_b   = (const float*)d_in[6];
    const float* gmm   = (const float*)d_in[7];
    const float* bet   = (const float*)d_in[8];
    const float* mu    = (const float*)d_in[9];
    const float* var   = (const float*)d_in[10];
    const float* alpha = (const float*)d_in[11];

    char* ws = (char*)d_ws;
    float*  qsum = (float*)(ws);                 // 16*5*2048 f32  = 655360 B
    __bf16* apS  = (__bf16*)(ws + 655360);       // 327680 bf16    = 655360 B
    __bf16* apV  = (__bf16*)(ws + 1310720);      // 65536 bf16     = 131072 B
    float*  qws  = (float*)(ws + 1441792);       // 1280 f32
    float*  qbs  = (float*)(ws + 1446912);       // 5 f32 (pad)
    float*  vsc  = (float*)(ws + 1446944);       // 256 f32
    float*  vsh  = (float*)(ws + 1447968);       // 256 f32

    prep1_kernel<<<NL + 1, 256, 0, stream>>>(q_w, q_b, v_b, gmm, bet, mu, var, qws, qbs, vsc, vsh);
    pack_kernel<<<(NSE + NVE) / 256, 256, 0, stream>>>(k_w, v_w, apS, apV);
    qsum_kernel<<<NBB * (TTT / 256), 256, 0, stream>>>(x, qws, qbs, qsum);

    hipFuncSetAttribute((const void*)main_kernel,
                        hipFuncAttributeMaxDynamicSharedMemorySize, (int)SMEM_BYTES);
    dim3 grid(TTT / 32, NBB);
    main_kernel<<<grid, 256, SMEM_BYTES, stream>>>(x, k_b, alpha, qsum, apS, apV, vsc, vsh,
                                                   (float*)d_out);
}
